// MixLinear_GEMM_49220325212290
// MI455X (gfx1250) — 16-bit and 8-bit matrix operand order repaired and checked by reading, compiled, not run on hardware
//
#include <hip/hip_runtime.h>
#include <hip/hip_bf16.h>


typedef __attribute__((ext_vector_type(2)))  int      v2i;
typedef __attribute__((ext_vector_type(4)))  int      v4i;
typedef __attribute__((ext_vector_type(8)))  int      v8i;
typedef __attribute__((ext_vector_type(8)))  float    v8f;
typedef __attribute__((ext_vector_type(8)))  _Float16 v8h;
typedef __attribute__((ext_vector_type(16))) _Float16 v16h;

#define KDIM 4096
#define NDIM 11008

__global__ __launch_bounds__(256) void zero_mask_kernel(int* __restrict__ mask, int K) {
    int k = blockIdx.x * 256 + threadIdx.x;
    if (k < K) mask[k] = 0;
}

__global__ __launch_bounds__(256) void find_outliers_kernel(
    const float* __restrict__ x, const float* __restrict__ sigma,
    int* __restrict__ mask, int M, int K, int rows_per_blk)
{
    int k = blockIdx.x * 256 + threadIdx.x;
    if (k >= K) return;
    float sig = sigma[0];
    int r0 = blockIdx.y * rows_per_blk;
    int r1 = r0 + rows_per_blk; if (r1 > M) r1 = M;
    int flag = 0;
    for (int m = r0; m < r1; ++m)
        flag |= (fabsf(x[(size_t)m * K + k]) > sig) ? 1 : 0;
    if (flag) atomicOr(&mask[k], 1);
}

__global__ __launch_bounds__(64) void tile_flag_kernel(
    const int* __restrict__ mask, int* __restrict__ tflag, int ntiles)
{
    int t = blockIdx.x * 64 + threadIdx.x;
    if (t >= ntiles) return;
    int f = 0;
    for (int i = 0; i < 64; ++i) f |= mask[t * 64 + i];
    tflag[t] = f;
}

__global__ __launch_bounds__(256) void quant_weight_kernel(
    const float* __restrict__ w, const int* __restrict__ mask,
    float* __restrict__ scale_col, signed char* __restrict__ qw,
    _Float16* __restrict__ wout, int K)
{
    __shared__ float red[256];
    int n = blockIdx.x;
    const float* wr = w + (size_t)n * K;
    float amax = 0.f;
    for (int k = threadIdx.x; k < K; k += 256) amax = fmaxf(amax, fabsf(wr[k]));
    red[threadIdx.x] = amax; __syncthreads();
    for (int s = 128; s > 0; s >>= 1) {
        if (threadIdx.x < s) red[threadIdx.x] = fmaxf(red[threadIdx.x], red[threadIdx.x + s]);
        __syncthreads();
    }
    float scale = fmaxf(red[0] * (1.f / 64.f), 1e-30f);
    if (threadIdx.x == 0) scale_col[n] = scale;
    float inv = 1.f / scale;
    for (int k = threadIdx.x; k < K; k += 256) {
        float q = rintf(wr[k] * inv);
        qw[(size_t)n * K + k]   = (signed char)q;
        wout[(size_t)n * K + k] = mask[k] ? (_Float16)q : (_Float16)0.f;
    }
}

__global__ __launch_bounds__(256) void quant_act_kernel(
    const float* __restrict__ x, const int* __restrict__ mask,
    float* __restrict__ x_scale, signed char* __restrict__ qx,
    _Float16* __restrict__ xout, int K)
{
    __shared__ float red[256];
    int m = blockIdx.x;
    const float* xr = x + (size_t)m * K;
    float amax = 0.f;
    for (int k = threadIdx.x; k < K; k += 256) {
        float v = mask[k] ? 0.f : fabsf(xr[k]);
        amax = fmaxf(amax, v);
    }
    red[threadIdx.x] = amax; __syncthreads();
    for (int s = 128; s > 0; s >>= 1) {
        if (threadIdx.x < s) red[threadIdx.x] = fmaxf(red[threadIdx.x], red[threadIdx.x + s]);
        __syncthreads();
    }
    float xs = fmaxf(red[0] * (1.f / 127.f), 1e-8f);
    if (threadIdx.x == 0) x_scale[m] = xs;
    float inv = 1.f / xs;
    for (int k = threadIdx.x; k < K; k += 256) {
        if (mask[k]) {
            qx[(size_t)m * K + k]   = 0;
            xout[(size_t)m * K + k] = (_Float16)xr[k];
        } else {
            qx[(size_t)m * K + k]   = (signed char)rintf(xr[k] * inv);
            xout[(size_t)m * K + k] = (_Float16)0.f;
        }
    }
}

__global__ __launch_bounds__(32) void mix_gemm_kernel(
    const signed char* __restrict__ qx, const signed char* __restrict__ qw,
    const _Float16* __restrict__ xout, const _Float16* __restrict__ wout,
    const float* __restrict__ x_scale, const float* __restrict__ scale_col,
    const int* __restrict__ tflag, float* __restrict__ y,
    int M, int N)
{
    const int lane = threadIdx.x;
    const int half = lane >> 4;
    const int l16  = lane & 15;
    const int M0 = blockIdx.y * 16;
    const int N0 = blockIdx.x * 64;

    const int    koffA = half * 8;
    const int    koffB = half * 16;
    const size_t rowA  = (size_t)(M0 + l16) * KDIM;
    size_t rowB[4];
#pragma unroll
    for (int t = 0; t < 4; ++t) rowB[t] = (size_t)(N0 + t * 16 + l16) * KDIM;

    auto loadA = [&](int kt) -> v8i {
        const signed char* ap = qx + rowA + kt + koffB;
        v4i a03 = *(const v4i*)(ap +  0);
        v4i a47 = *(const v4i*)(ap + 32);
        return (v8i){a03.x, a03.y, a03.z, a03.w, a47.x, a47.y, a47.z, a47.w};
    };
    auto loadB = [&](int t, int kt) -> v8i {
        const signed char* bp = qw + rowB[t] + kt + koffB;
        v4i b03 = *(const v4i*)(bp +  0);
        v4i b47 = *(const v4i*)(bp + 32);
        return (v8i){b03.x, b03.y, b03.z, b03.w, b47.x, b47.y, b47.z, b47.w};
    };

    v8i acc[4];
#pragma unroll
    for (int t = 0; t < 4; ++t)
#pragma unroll
        for (int r = 0; r < 8; ++r) acc[t][r] = 0;

    v8i A0  = loadA(0);
    v8i B00 = loadB(0, 0), B01 = loadB(1, 0), B02 = loadB(2, 0), B03 = loadB(3, 0);

#pragma unroll 1
    for (int kt = 0; kt < KDIM; kt += 128) {
        const int k1 = kt + 64;
        v8i A1  = loadA(k1);
        v8i B10 = loadB(0, k1), B11 = loadB(1, k1), B12 = loadB(2, k1), B13 = loadB(3, k1);

        acc[0] = __builtin_amdgcn_wmma_i32_16x16x64_iu8(true, A0, true, B00, acc[0], false, false);
        acc[1] = __builtin_amdgcn_wmma_i32_16x16x64_iu8(true, A0, true, B01, acc[1], false, false);
        acc[2] = __builtin_amdgcn_wmma_i32_16x16x64_iu8(true, A0, true, B02, acc[2], false, false);
        acc[3] = __builtin_amdgcn_wmma_i32_16x16x64_iu8(true, A0, true, B03, acc[3], false, false);

        const int k2 = (kt + 128 < KDIM) ? kt + 128 : 0;
        A0  = loadA(k2);
        B00 = loadB(0, k2); B01 = loadB(1, k2); B02 = loadB(2, k2); B03 = loadB(3, k2);

        acc[0] = __builtin_amdgcn_wmma_i32_16x16x64_iu8(true, A1, true, B10, acc[0], false, false);
        acc[1] = __builtin_amdgcn_wmma_i32_16x16x64_iu8(true, A1, true, B11, acc[1], false, false);
        acc[2] = __builtin_amdgcn_wmma_i32_16x16x64_iu8(true, A1, true, B12, acc[2], false, false);
        acc[3] = __builtin_amdgcn_wmma_i32_16x16x64_iu8(true, A1, true, B13, acc[3], false, false);
    }

    v8f accf[4];
#pragma unroll
    for (int t = 0; t < 4; ++t)
#pragma unroll
        for (int r = 0; r < 8; ++r) accf[t][r] = 0.f;

    for (int kt = 0; kt < KDIM; kt += 64) {
        if (!tflag[kt >> 6]) continue;

        const _Float16* xp = xout + rowA + kt + koffA;
        v8h a0lo = *(const v8h*)(xp +  0);
        v8h a0hi = *(const v8h*)(xp + 16);
        v8h a1lo = *(const v8h*)(xp + 32);
        v8h a1hi = *(const v8h*)(xp + 48);
        v16h Af0, Af1;
#pragma unroll
        for (int i = 0; i < 8; ++i) {
            Af0[i] = a0lo[i]; Af0[8 + i] = a0hi[i];
            Af1[i] = a1lo[i]; Af1[8 + i] = a1hi[i];
        }

#pragma unroll
        for (int t = 0; t < 4; ++t) {
            const _Float16* wp = wout + rowB[t] + kt + koffA;
            v8h b0lo = *(const v8h*)(wp +  0), b0hi = *(const v8h*)(wp + 16);
            v8h b1lo = *(const v8h*)(wp + 32), b1hi = *(const v8h*)(wp + 48);
            v16h Bf0 = __builtin_shufflevector(b0lo, b0hi, 0,1,2,3,4,5,6,7,8,9,10,11,12,13,14,15);
            v16h Bf1 = __builtin_shufflevector(b1lo, b1hi, 0,1,2,3,4,5,6,7,8,9,10,11,12,13,14,15);
            accf[t] = __builtin_amdgcn_wmma_f32_16x16x32_f16(
                false, Af0, false, Bf0, (short)0, accf[t], false, false);
            accf[t] = __builtin_amdgcn_wmma_f32_16x16x32_f16(
                false, Af1, false, Bf1, (short)0, accf[t], false, false);
        }
    }

    float xs[8];
#pragma unroll
    for (int r = 0; r < 8; ++r) xs[r] = x_scale[M0 + r + half * 8];
#pragma unroll
    for (int t = 0; t < 4; ++t) {
        const int n = N0 + t * 16 + l16;
        const float sc = scale_col[n];
#pragma unroll
        for (int r = 0; r < 8; ++r) {
            const int m = M0 + r + half * 8;
            y[(size_t)m * N + n] = sc * ((float)acc[t][r] * xs[r] + accf[t][r]);
        }
    }
}

extern "C" void kernel_launch(void* const* d_in, const int* in_sizes, int n_in,
                              void* d_out, int out_size, void* d_ws, size_t ws_size,
                              hipStream_t stream)
{
    const float* x     = (const float*)d_in[0];
    const float* w     = (const float*)d_in[1];
    const float* sigma = (const float*)d_in[2];

    const int K = KDIM;
    const int N = NDIM;
    const int M = in_sizes[0] / K;

    char* ws = (char*)d_ws;
    size_t off = 0;
    auto take = [&](size_t bytes) -> char* {
        char* p = ws + off;
        off += (bytes + 255) & ~(size_t)255;
        return p;
    };
    int*         mask      = (int*)        take((size_t)K * sizeof(int));
    int*         tflag     = (int*)        take((size_t)(K / 64) * sizeof(int));
    float*       scale_col = (float*)      take((size_t)N * sizeof(float));
    float*       x_scale   = (float*)      take((size_t)M * sizeof(float));
    signed char* qw        = (signed char*)take((size_t)N * K);
    signed char* qx        = (signed char*)take((size_t)M * K);
    _Float16*    wout      = (_Float16*)   take((size_t)N * K * 2);
    _Float16*    xout      = (_Float16*)   take((size_t)M * K * 2);

    zero_mask_kernel<<<(K + 255) / 256, 256, 0, stream>>>(mask, K);

    const int row_chunks = 64;
    const int rows_per_blk = (M + row_chunks - 1) / row_chunks;
    dim3 gmask((K + 255) / 256, row_chunks);
    find_outliers_kernel<<<gmask, 256, 0, stream>>>(x, sigma, mask, M, K, rows_per_blk);

    tile_flag_kernel<<<(K / 64 + 63) / 64, 64, 0, stream>>>(mask, tflag, K / 64);

    quant_weight_kernel<<<N, 256, 0, stream>>>(w, mask, scale_col, qw, wout, K);

    quant_act_kernel<<<M, 256, 0, stream>>>(x, mask, x_scale, qx, xout, K);

    dim3 gg(N / 64, M / 16);
    mix_gemm_kernel<<<gg, 32, 0, stream>>>(qx, qw, xout, wout, x_scale, scale_col,
                                           tflag, (float*)d_out, M, N);
}
